// WhatsnetIM_11063835755287
// MI455X (gfx1250) — matrix operand order repaired and checked by reading, not compiled, not run on hardware
//
#include <hip/hip_runtime.h>
#include <hip/hip_bf16.h>

typedef _Float16 v16h __attribute__((ext_vector_type(16)));
typedef _Float16 v8h  __attribute__((ext_vector_type(8)));
typedef _Float16 v4h  __attribute__((ext_vector_type(4)));
typedef float    v8f  __attribute__((ext_vector_type(8)));

// Row strides chosen to rotate LDS banks by 4 per row (conflict-free column access)
// while keeping 16-byte alignment: 136 halves = 272 B = 17*16; 132 floats = 528 B = 33*16.
#define AS 136   // f16 activation row stride (halves)
#define XS 132   // f32 row stride (floats)

// ---------------------------------------------------------------------------
// WMMA fragment helpers (wave32, V_WMMA_F32_16X16X32_F16)
// A (16x32 f16): lane L row M=L&15; halves 0..7 = K[kb*32+(L>>4)*8+j], halves 8..15 = +16
// B (32x16 f16): lane L col N=L&15; halves 0..7 = K[kb*32+(L>>4)*8+j], halves 8..15 = +16
// C/D (16x16 f32): lane L, VGPR i -> row = i+(L>>4)*8, col = L&15
// ---------------------------------------------------------------------------
__device__ __forceinline__ v16h cat8(v8h lo, v8h hi) {
  return __builtin_shufflevector(lo, hi, 0,1,2,3,4,5,6,7,8,9,10,11,12,13,14,15);
}
__device__ __forceinline__ v16h load_a_frag(const _Float16* base, int lane, int kb) {
  int row = lane & 15;
  int ko  = kb * 32 + ((lane >> 4) << 3);
  const _Float16* p = base + row * AS + ko;
  v8h c0 = *(const v8h*)p;
  v8h c1 = *(const v8h*)(p + 16);
  return cat8(c0, c1);
}
__device__ __forceinline__ v16h load_b_frag(const _Float16* base, int lane, int kb) {
  int row = lane & 15;
  int ko  = kb * 32 + ((lane >> 4) << 3);
  const _Float16* p = base + row * AS + ko;
  v8h c0 = *(const v8h*)p;
  v8h c1 = *(const v8h*)(p + 16);
  return cat8(c0, c1);
}
__device__ __forceinline__ v8f wmma_f16(v16h a, v16h b, v8f c) {
  return __builtin_amdgcn_wmma_f32_16x16x32_f16(false, a, false, b, (short)0, c, false, false);
}
__device__ __forceinline__ float4 ld4(const float* p) { return *(const float4*)p; }

// ---------------------------------------------------------------------------
// Prep kernels (once per launch; tiny)
// ---------------------------------------------------------------------------
__global__ void prep_weights(const float* W0, const float* W1, const float* W2, const float* W3,
                             const float* W4, const float* W5, const float* W6, const float* W7,
                             _Float16* WT) {
  const float* Ws[8] = {W0, W1, W2, W3, W4, W5, W6, W7};
  int m = blockIdx.y;
  int t = blockIdx.x * blockDim.x + threadIdx.x;   // 0..16383
  int k = t >> 7, n = t & 127;
  WT[m * 16384 + n * 128 + k] = (_Float16)Ws[m][k * 128 + n];  // transposed f16
}

__global__ void prep_iq(const float* __restrict__ I, const float* __restrict__ Wq0,
                        const float* __restrict__ bq0, float* __restrict__ Iq) {
  int t = threadIdx.x;           // 512 threads: m = t>>7, d = t&127
  int m = t >> 7, d = t & 127;
  float acc = bq0[d];
  for (int k = 0; k < 128; ++k) acc += I[m * 128 + k] * Wq0[k * 128 + d];
  Iq[t] = acc;
}

// ---------------------------------------------------------------------------
// Main kernel: 8 hyperedges per workgroup (8 waves), LDS-staged padded weights
// ---------------------------------------------------------------------------
__global__ __launch_bounds__(256, 1) void isab_edge_kernel(
    const float* __restrict__ vfeat, const int* __restrict__ midx_g,
    const _Float16* __restrict__ WT, const float* __restrict__ IqG,
    const float* __restrict__ bk0, const float* __restrict__ bv0, const float* __restrict__ bo0,
    const float* __restrict__ bq1, const float* __restrict__ bk1, const float* __restrict__ bv1,
    const float* __restrict__ bo1,
    const float* __restrict__ Wout, const float* __restrict__ bout,
    float* __restrict__ logits, int E) {
  __shared__ _Float16 Wst[2 * 128 * AS];   // 68 KB staged weights (padded rows)
  __shared__ _Float16 aH[8][16 * AS];      // v (f16)          -> later O1 (f16)
  __shared__ _Float16 bH[8][16 * AS];      // K0 (f16)         -> later Hset padded
  __shared__ _Float16 cH[8][16 * AS];      // V0 (f16)         -> O_mab0 padded (in place)
  __shared__ float    X[8][16 * XS];       // Q1 -> O1 -> out (f32, in place)
  __shared__ float    sc[8][256];
  __shared__ float    Of[8][4 * XS];
  __shared__ float    K1s[8][4 * XS];
  __shared__ float    V1s[8][4 * XS];
  __shared__ float    IqS[4 * XS];
  __shared__ int      mi[8][16];

  const int tid   = threadIdx.x;
  const int lane  = tid & 31;
  const int w     = tid >> 5;              // wave id == edge slot
  const int col16 = lane & 15;
  const int rhalf = (lane >> 4) * 8;
  const float scale = 0.08838834764831845f;   // 1/sqrt(128): ref scales by sqrt(dim_V)

  int eg = blockIdx.x * 8 + w;
  const bool active = eg < E;
  if (!active) eg = E - 1;

  // staging: global loads issued BEFORE the leading barrier (latency overlapped
  // with previous phase tail + barrier wait); then padded LDS rows stored.
  auto stage1 = [&](const _Float16* s0) {
    v8h r0[8];
    const v8h* a0 = (const v8h*)s0;
#pragma unroll
    for (int i = 0; i < 8; ++i) r0[i] = a0[i * 256 + tid];
    __syncthreads();                               // previous users of Wst done
#pragma unroll
    for (int i = 0; i < 8; ++i) {
      int c8 = i * 256 + tid;                      // 16-byte chunk id
      int row = c8 >> 4, col = (c8 & 15) << 3;
      *(v8h*)(&Wst[row * AS + col]) = r0[i];
    }
    __syncthreads();
  };
  auto stage2 = [&](const _Float16* s0, const _Float16* s1) {
    v8h r0[8], r1[8];
    const v8h* a0 = (const v8h*)s0;
    const v8h* a1 = (const v8h*)s1;
#pragma unroll
    for (int i = 0; i < 8; ++i) { r0[i] = a0[i * 256 + tid]; r1[i] = a1[i * 256 + tid]; }
    __syncthreads();
#pragma unroll
    for (int i = 0; i < 8; ++i) {
      int c8 = i * 256 + tid;
      int row = c8 >> 4, col = (c8 & 15) << 3;
      *(v8h*)(&Wst[row * AS + col])            = r0[i];
      *(v8h*)(&Wst[128 * AS + row * AS + col]) = r1[i];
    }
    __syncthreads();
  };

  // ---- init
  if (lane < 16) mi[w][lane] = midx_g[eg * 16 + lane];
  for (int t = tid; t < 512; t += 256) IqS[(t >> 7) * XS + (t & 127)] = IqG[t];
  __syncthreads();

  // ---- gather 16 member rows per edge, f32 -> f16 into aH[w]
  for (int s = 0; s < 16; ++s) {
    const float4 f = ((const float4*)(vfeat + (size_t)mi[w][s] * 128))[lane];
    v4h h4 = {(_Float16)f.x, (_Float16)f.y, (_Float16)f.z, (_Float16)f.w};
    *(v4h*)(&aH[w][s * AS + lane * 4]) = h4;
  }

  // ---- MAB0: K0 = v@Wk0+bk0 -> bH;  V0 = v@Wv0+bv0 -> cH
  v16h afv[4];
#pragma unroll
  for (int kb = 0; kb < 4; ++kb) afv[kb] = load_a_frag(&aH[w][0], lane, kb);
  stage2(WT + 1 * 16384, WT + 2 * 16384);
  for (int nb = 0; nb < 8; ++nb) {
    v8f accK = {}; v8f accV = {};
#pragma unroll
    for (int kb = 0; kb < 4; ++kb) {
      accK = wmma_f16(afv[kb], load_b_frag(Wst + nb * 16 * AS, lane, kb), accK);
      accV = wmma_f16(afv[kb], load_b_frag(Wst + 128 * AS + nb * 16 * AS, lane, kb), accV);
    }
    int n = nb * 16 + col16;
    float bk = bk0[n], bv = bv0[n];
    for (int i = 0; i < 8; ++i) {
      int r = i + rhalf;
      bH[w][r * AS + n] = (_Float16)(accK[i] + bk);
      cH[w][r * AS + n] = (_Float16)(accV[i] + bv);
    }
  }

  // ---- MAB0 scores: sc[(h*4+m)*16+s] = (Iq[m] . K0[s])_head_h * scale
  for (int t = 0; t < 8; ++t) {
    int e = lane + 32 * t; int s = e & 15; int r = e >> 4;
    int h = r >> 2, m = r & 3, d0 = h * 32;
    const float* iq = &IqS[m * XS + d0];
    const v8h* kv = (const v8h*)(&bH[w][s * AS + d0]);
    float acc = 0.f;
#pragma unroll
    for (int j = 0; j < 4; ++j) {
      v8h kk = kv[j];
      float4 a0 = ld4(iq + j * 8), a1 = ld4(iq + j * 8 + 4);
      acc += a0.x * (float)kk[0] + a0.y * (float)kk[1] + a0.z * (float)kk[2] + a0.w * (float)kk[3]
           + a1.x * (float)kk[4] + a1.y * (float)kk[5] + a1.z * (float)kk[6] + a1.w * (float)kk[7];
    }
    sc[w][r * 16 + s] = acc * scale;
  }
  if (lane < 16) {                       // softmax over 16 keys, one (h,m) row per lane
    float mx = -1e30f;
    for (int s = 0; s < 16; ++s) mx = fmaxf(mx, sc[w][lane * 16 + s]);
    float sum = 0.f;
    for (int s = 0; s < 16; ++s) { float ev = __expf(sc[w][lane * 16 + s] - mx); sc[w][lane * 16 + s] = ev; sum += ev; }
    float inv = 1.f / sum;
    for (int s = 0; s < 16; ++s) sc[w][lane * 16 + s] *= inv;
  }

  // ---- O = Iq + A@V0 : column-owned (lane owns 4 contiguous d), cH overwritten
  //      in place with zero-padded f16(O); f32 rows 0..3 -> Of. Also pad bH rows.
  {
    int dL = lane * 4; int h = lane >> 3;
    float4 vv[16];
#pragma unroll
    for (int s = 0; s < 16; ++s) {
      v4h hv = *(const v4h*)(&cH[w][s * AS + dL]);
      vv[s] = make_float4((float)hv[0], (float)hv[1], (float)hv[2], (float)hv[3]);
    }
#pragma unroll
    for (int m = 0; m < 4; ++m) {
      float4 om = ld4(&IqS[m * XS + dL]);
      int r = h * 4 + m;
#pragma unroll
      for (int s = 0; s < 16; ++s) {
        float a = sc[w][r * 16 + s];
        om.x += a * vv[s].x; om.y += a * vv[s].y; om.z += a * vv[s].z; om.w += a * vv[s].w;
      }
      *(float4*)(&Of[w][m * XS + dL]) = om;
      v4h oh = {(_Float16)om.x, (_Float16)om.y, (_Float16)om.z, (_Float16)om.w};
      *(v4h*)(&cH[w][m * AS + dL]) = oh;
    }
    v4h z4 = {(_Float16)0.f, (_Float16)0.f, (_Float16)0.f, (_Float16)0.f};
#pragma unroll
    for (int m = 4; m < 16; ++m) {
      *(v4h*)(&cH[w][m * AS + dL]) = z4;
      *(v4h*)(&bH[w][m * AS + dL]) = z4;
    }
  }

  // ---- paired stage: slot0 = Wo0, slot1 = Wq1 (two gemm phases, one staging)
  v16h afo[4];
#pragma unroll
  for (int kb = 0; kb < 4; ++kb) afo[kb] = load_a_frag(&cH[w][0], lane, kb);
  stage2(WT + 3 * 16384, WT + 4 * 16384);

  // Hset = O + relu(O@Wo0+bo0) -> bH rows 0..3 (f16, padded)
  for (int nb = 0; nb < 8; ++nb) {
    v8f acc = {};
#pragma unroll
    for (int kb = 0; kb < 4; ++kb)
      acc = wmma_f16(afo[kb], load_b_frag(Wst + nb * 16 * AS, lane, kb), acc);
    int n = nb * 16 + col16;
    float bo = bo0[n];
    if (lane < 16) {
      for (int i = 0; i < 4; ++i) {
        float hv = Of[w][i * XS + n] + fmaxf(acc[i] + bo, 0.f);
        bH[w][i * AS + n] = (_Float16)hv;
      }
    }
  }

  // MAB1: Q1 = v@Wq1+bq1 -> X (f32)   (slot1 of the same staging)
  for (int nb = 0; nb < 8; ++nb) {
    v8f acc = {};
#pragma unroll
    for (int kb = 0; kb < 4; ++kb)
      acc = wmma_f16(afv[kb], load_b_frag(Wst + 128 * AS + nb * 16 * AS, lane, kb), acc);
    int n = nb * 16 + col16; float bq = bq1[n];
    for (int i = 0; i < 8; ++i) X[w][(i + rhalf) * XS + n] = acc[i] + bq;
  }

  // ---- K1 = Hset@Wk1+bk1, V1 = Hset@Wv1+bv1 (rows 0..3 valid)
  v16h afh[4];
#pragma unroll
  for (int kb = 0; kb < 4; ++kb) afh[kb] = load_a_frag(&bH[w][0], lane, kb);
  stage2(WT + 5 * 16384, WT + 6 * 16384);
  for (int nb = 0; nb < 8; ++nb) {
    v8f accK = {}; v8f accV = {};
#pragma unroll
    for (int kb = 0; kb < 4; ++kb) {
      accK = wmma_f16(afh[kb], load_b_frag(Wst + nb * 16 * AS, lane, kb), accK);
      accV = wmma_f16(afh[kb], load_b_frag(Wst + 128 * AS + nb * 16 * AS, lane, kb), accV);
    }
    int n = nb * 16 + col16;
    if (lane < 16) {
      float bk = bk1[n], bv = bv1[n];
      for (int i = 0; i < 4; ++i) { K1s[w][i * XS + n] = accK[i] + bk; V1s[w][i * XS + n] = accV[i] + bv; }
    }
  }

  // ---- MAB1 scores: sc[(h*16+q)*4+k]
  for (int t = 0; t < 8; ++t) {
    int e = lane + 32 * t; int k = e & 3; int q = (e >> 2) & 15; int h = e >> 6;
    int d0 = h * 32;
    const float* xq = &X[w][q * XS + d0];
    const float* kk = &K1s[w][k * XS + d0];
    float4 a4 = {0.f, 0.f, 0.f, 0.f};
#pragma unroll
    for (int j = 0; j < 8; ++j) {
      float4 xa = ld4(xq + j * 4), ka = ld4(kk + j * 4);
      a4.x += xa.x * ka.x; a4.y += xa.y * ka.y; a4.z += xa.z * ka.z; a4.w += xa.w * ka.w;
    }
    sc[w][e] = (a4.x + a4.y + a4.z + a4.w) * scale;
  }
  for (int t = 0; t < 2; ++t) {          // softmax over 4 keys, 64 groups
    int g = lane + 32 * t;
    float a0 = sc[w][g * 4], a1 = sc[w][g * 4 + 1], a2 = sc[w][g * 4 + 2], a3 = sc[w][g * 4 + 3];
    float mx = fmaxf(fmaxf(a0, a1), fmaxf(a2, a3));
    float x0 = __expf(a0 - mx), x1 = __expf(a1 - mx), x2 = __expf(a2 - mx), x3 = __expf(a3 - mx);
    float inv = 1.f / (x0 + x1 + x2 + x3);
    sc[w][g * 4] = x0 * inv; sc[w][g * 4 + 1] = x1 * inv;
    sc[w][g * 4 + 2] = x2 * inv; sc[w][g * 4 + 3] = x3 * inv;
  }

  // ---- O1 = Q1 + A@V1 (in-place X, column-owned); aH = f16(O1)
  {
    int dL = lane * 4; int h = lane >> 3;
#pragma unroll
    for (int q = 0; q < 16; ++q) {
      float4 x = ld4(&X[w][q * XS + dL]);
      int sb = (h * 16 + q) * 4;
#pragma unroll
      for (int k = 0; k < 4; ++k) {
        float a = sc[w][sb + k];
        float4 vk = ld4(&V1s[w][k * XS + dL]);
        x.x += a * vk.x; x.y += a * vk.y; x.z += a * vk.z; x.w += a * vk.w;
      }
      *(float4*)(&X[w][q * XS + dL]) = x;
      v4h oh = {(_Float16)x.x, (_Float16)x.y, (_Float16)x.z, (_Float16)x.w};
      *(v4h*)(&aH[w][q * AS + dL]) = oh;
    }
  }

  // ---- out = O1 + relu(O1@Wo1+bo1) (in-place X)
  v16h afx[4];
#pragma unroll
  for (int kb = 0; kb < 4; ++kb) afx[kb] = load_a_frag(&aH[w][0], lane, kb);
  stage1(WT + 7 * 16384);
  for (int nb = 0; nb < 8; ++nb) {
    v8f acc = {};
#pragma unroll
    for (int kb = 0; kb < 4; ++kb)
      acc = wmma_f16(afx[kb], load_b_frag(Wst + nb * 16 * AS, lane, kb), acc);
    int n = nb * 16 + col16; float bo = bo1[n];
    for (int i = 0; i < 8; ++i) { int r = i + rhalf; X[w][r * XS + n] += fmaxf(acc[i] + bo, 0.f); }
  }

  // ---- logits: stage Wout^T (f32) into Wst space, then vector dot
  __syncthreads();                        // all waves done reading Wst as weights
  float* WoutF = (float*)Wst;             // reuse staging space: [6][XS] f32
  for (int t = tid; t < 768; t += 256) {
    int c = t >> 7, d = t & 127;
    WoutF[c * XS + d] = Wout[d * 6 + c];
  }
  __syncthreads();

  if (active) {
#pragma unroll
    for (int t = 0; t < 3; ++t) {
      int p = lane + 32 * t; int s = p / 6; int c = p % 6;
      const float* xr = &X[w][s * XS];
      const float* wr = &WoutF[c * XS];
      float4 a4 = {0.f, 0.f, 0.f, 0.f};
#pragma unroll
      for (int j = 0; j < 32; ++j) {
        float4 xa = ld4(xr + j * 4), wa = ld4(wr + j * 4);
        a4.x += xa.x * wa.x; a4.y += xa.y * wa.y; a4.z += xa.z * wa.z; a4.w += xa.w * wa.w;
      }
      logits[((size_t)eg * 16 + s) * 6 + c] = bout[c] + a4.x + a4.y + a4.z + a4.w;
    }
  }
}

// ---------------------------------------------------------------------------
extern "C" void kernel_launch(void* const* d_in, const int* in_sizes, int n_in,
                              void* d_out, int out_size, void* d_ws, size_t ws_size,
                              hipStream_t stream) {
  const float* vfeat = (const float*)d_in[0];
  const int*   midx  = (const int*)d_in[1];
  const int*   labels= (const int*)d_in[2];
  const float* I     = (const float*)d_in[3];
  const float* Wout  = (const float*)d_in[4];
  const float* bout  = (const float*)d_in[5];
  const float* m0Wq = (const float*)d_in[6],  *m0bq = (const float*)d_in[7];
  const float* m0Wk = (const float*)d_in[8],  *m0bk = (const float*)d_in[9];
  const float* m0Wv = (const float*)d_in[10], *m0bv = (const float*)d_in[11];
  const float* m0Wo = (const float*)d_in[12], *m0bo = (const float*)d_in[13];
  const float* m1Wq = (const float*)d_in[14], *m1bq = (const float*)d_in[15];
  const float* m1Wk = (const float*)d_in[16], *m1bk = (const float*)d_in[17];
  const float* m1Wv = (const float*)d_in[18], *m1bv = (const float*)d_in[19];
  const float* m1Wo = (const float*)d_in[20], *m1bo = (const float*)d_in[21];

  const int E  = in_sizes[1] / 16;           // member_idx is [E,16]
  const int ES = E * 16;

  _Float16* WT = (_Float16*)d_ws;            // 8 * 128*128 f16 = 256 KB (unpadded)
  float*    Iq = (float*)((char*)d_ws + 8 * 16384 * sizeof(_Float16));  // 4*128 f32

  prep_weights<<<dim3(64, 8), 256, 0, stream>>>(m0Wq, m0Wk, m0Wv, m0Wo,
                                                m1Wq, m1Wk, m1Wv, m1Wo, WT);
  prep_iq<<<1, 512, 0, stream>>>(I, m0Wq, m0bq, Iq);

  // labels passthrough (second tuple output, raw int32 bits)
  int* labOut = ((int*)d_out) + (size_t)ES * 6;
  hipMemcpyAsync(labOut, labels, (size_t)ES * sizeof(int), hipMemcpyDeviceToDevice, stream);

  const int nBlocks = (E + 7) / 8;
  isab_edge_kernel<<<nBlocks, 256, 0, stream>>>(vfeat, midx, WT, Iq,
                                                m0bk, m0bv, m0bo,
                                                m1bq, m1bk, m1bv, m1bo,
                                                Wout, bout, (float*)d_out, E);
}
